// TILTRegression_68728066670857
// MI455X (gfx1250) — compile-verified, matrix operand order checked by reading, not run on hardware
//
#include <hip/hip_runtime.h>
#include <stdint.h>

// ---------------------------------------------------------------------------
// Types for CDNA5 WMMA (wave32): A/B = 16 bf16 per lane, C/D = 8 f32 per lane
// ---------------------------------------------------------------------------
typedef __attribute__((ext_vector_type(16))) __bf16 v16bf;
typedef __attribute__((ext_vector_type(8)))  float  v8f;
typedef __attribute__((ext_vector_type(4)))  int    i4vec;

#define NEGV -1000000000.0f

#if __has_builtin(__builtin_amdgcn_global_load_async_to_lds_b128)
#define HAS_ASYNC_LDS 1
#endif

__device__ __forceinline__ uint16_t f2bfu(float f) {
  uint32_t u = __float_as_uint(f);
  uint32_t r = u + 0x7FFFu + ((u >> 16) & 1u);   // round-to-nearest-even
  return (uint16_t)(r >> 16);
}
__device__ __forceinline__ __bf16 u2bf(uint16_t u) {
  __bf16 b; __builtin_memcpy(&b, &u, 2); return b;
}

__device__ __forceinline__ v8f wmma_bf16(v16bf a, v16bf b, v8f c) {
  // (neg_a, A, neg_b, B, c_mod, C, reuse_a, reuse_b)
  return __builtin_amdgcn_wmma_f32_16x16x32_bf16(false, a, false, b, (short)0, c,
                                                 false, false);
}

// 16-byte copy global(bf16) -> LDS, async (ASYNCcnt path) when available.
__device__ __forceinline__ void cp16_g2l(const uint16_t* g, uint16_t* l) {
#ifdef HAS_ASYNC_LDS
  __builtin_amdgcn_global_load_async_to_lds_b128(
      (__attribute__((address_space(1))) i4vec*)g,
      (__attribute__((address_space(3))) i4vec*)l, 0, 0);
#else
  *(uint4*)l = *(const uint4*)g;
#endif
}
__device__ __forceinline__ void async_drain() {
#ifdef HAS_ASYNC_LDS
#if __has_builtin(__builtin_amdgcn_s_wait_asynccnt)
  __builtin_amdgcn_s_wait_asynccnt(0);
#else
  asm volatile("s_wait_asynccnt 0" ::: "memory");
#endif
#endif
}

// Load a 16x32 bf16 A-layout fragment from a row-major [rows x ld] u16 matrix.
// ISA lane layout (16-bit A 16x32): lanes 0-15 row=lr, e0..7=K0..7, e8..15=K16..23;
// lanes 16-31 row=lr, e0..7=K8..15, e8..15=K24..31.
__device__ __forceinline__ v16bf ld_frag(const uint16_t* base, int ld, int row0,
                                         int k0, int lane) {
  const int lr = lane & 15, half = lane >> 4;
  const uint16_t* p = base + (size_t)(row0 + lr) * ld + k0 + half * 8;
  v16bf f;
#pragma unroll
  for (int i = 0; i < 8; ++i) {
    f[i]     = u2bf(p[i]);
    f[i + 8] = u2bf(p[i + 16]);
  }
  return f;
}

// B-fragment for P@V where V sits row-major [key x 64] in LDS: element i maps
// to key = kb0 + kpat(i), dim = n0 + lr  (transposed access, stride 64).
__device__ __forceinline__ v16bf ld_frag_bT(const uint16_t* vs, int kb0, int n0,
                                            int lane) {
  const int lr = lane & 15, half = lane >> 4;
  const uint16_t* p = vs + (size_t)(kb0 + half * 8) * 64 + n0 + lr;
  v16bf f;
#pragma unroll
  for (int i = 0; i < 8; ++i) {
    f[i]     = u2bf(p[(size_t)i * 64]);
    f[i + 8] = u2bf(p[(size_t)(i + 16) * 64]);
  }
  return f;
}

// ---------------------------------------------------------------------------
// Embedding gather + LayerNorm  (one block per token, D = 768 = 3*256)
// ---------------------------------------------------------------------------
__global__ __launch_bounds__(256) void embed_ln_kernel(
    const int* __restrict__ ids, const float* __restrict__ wemb,
    const float* __restrict__ pemb, const float* __restrict__ temb,
    const float* __restrict__ gs, const float* __restrict__ gb,
    float* __restrict__ h, uint16_t* __restrict__ hb, int S, int D) {
  const int tok = blockIdx.x;
  const int t   = tok % S;
  const int tid = threadIdx.x;
  const int id  = ids[tok];
  const float* we = wemb + (size_t)id * D;
  const float* pe = pemb + (size_t)(t + 2) * D;
  float x[3]; float sum = 0.f, sq = 0.f;
#pragma unroll
  for (int j = 0; j < 3; ++j) {
    const int d = tid + j * 256;
    x[j] = we[d] + pe[d] + temb[d];
    sum += x[j]; sq += x[j] * x[j];
  }
  __shared__ float rs[256], rq[256];
  rs[tid] = sum; rq[tid] = sq; __syncthreads();
  for (int s2 = 128; s2 > 0; s2 >>= 1) {
    if (tid < s2) { rs[tid] += rs[tid + s2]; rq[tid] += rq[tid + s2]; }
    __syncthreads();
  }
  const float mean = rs[0] / D;
  const float var  = rq[0] / D - mean * mean;
  const float inv  = rsqrtf(var + 1e-5f);
#pragma unroll
  for (int j = 0; j < 3; ++j) {
    const int d = tid + j * 256;
    const float y = (x[j] - mean) * inv * gs[d] + gb[d];
    h[(size_t)tok * D + d]  = y;
    hb[(size_t)tok * D + d] = f2bfu(y);
  }
}

// ---------------------------------------------------------------------------
// Residual add + LayerNorm (one block per token)
// ---------------------------------------------------------------------------
__global__ __launch_bounds__(256) void add_ln_kernel(
    const float* __restrict__ a, const float* __restrict__ tin,
    const float* __restrict__ gs, const float* __restrict__ gb,
    float* __restrict__ h, uint16_t* __restrict__ hb, int D) {
  const int tok = blockIdx.x;
  const int tid = threadIdx.x;
  float x[3]; float sum = 0.f, sq = 0.f;
#pragma unroll
  for (int j = 0; j < 3; ++j) {
    const int d = tid + j * 256;
    const size_t idx = (size_t)tok * D + d;
    x[j] = a[idx] + tin[idx];
    sum += x[j]; sq += x[j] * x[j];
  }
  __shared__ float rs[256], rq[256];
  rs[tid] = sum; rq[tid] = sq; __syncthreads();
  for (int s2 = 128; s2 > 0; s2 >>= 1) {
    if (tid < s2) { rs[tid] += rs[tid + s2]; rq[tid] += rq[tid + s2]; }
    __syncthreads();
  }
  const float mean = rs[0] / D;
  const float var  = rq[0] / D - mean * mean;
  const float inv  = rsqrtf(var + 1e-5f);
#pragma unroll
  for (int j = 0; j < 3; ++j) {
    const int d = tid + j * 256;
    const size_t idx = (size_t)tok * D + d;
    const float y = (x[j] - mean) * inv * gs[d] + gb[d];
    h[idx]  = y;
    hb[idx] = f2bfu(y);
  }
}

// ---------------------------------------------------------------------------
// Generic bf16 WMMA GEMM: C[M,N] = act(A[M,K] @ W[K,N] + bias)
// A: row-major bf16 (M x K). Wt: transposed weights, row-major bf16 (N x K).
// Workgroup tile 128x64: 8 waves in 4x2, each wave 2x2 frags of 16x16.
// act: 0 = none, 1 = exact GELU. Either f32 (C) or bf16 (Cb) output may be null.
// ---------------------------------------------------------------------------
__global__ __launch_bounds__(256) void gemm_kernel(
    const uint16_t* __restrict__ A, const uint16_t* __restrict__ Wt,
    const float* __restrict__ bias, float* __restrict__ C,
    uint16_t* __restrict__ Cb, int M, int N, int K, int act) {
  const int lane = threadIdx.x & 31;
  const int wave = threadIdx.x >> 5;
  const int m0 = blockIdx.y * 128 + (wave >> 1) * 32;
  const int n0 = blockIdx.x * 64 + (wave & 1) * 32;

  v8f acc[2][2] = {};
  for (int k0 = 0; k0 < K; k0 += 32) {
    v16bf a0 = ld_frag(A, K, m0, k0, lane);
    v16bf a1 = ld_frag(A, K, m0 + 16, k0, lane);
    v16bf b0 = ld_frag(Wt, K, n0, k0, lane);
    v16bf b1 = ld_frag(Wt, K, n0 + 16, k0, lane);
    acc[0][0] = wmma_bf16(a0, b0, acc[0][0]);
    acc[0][1] = wmma_bf16(a0, b1, acc[0][1]);
    acc[1][0] = wmma_bf16(a1, b0, acc[1][0]);
    acc[1][1] = wmma_bf16(a1, b1, acc[1][1]);
  }

  const int lr = lane & 15, half = lane >> 4;
#pragma unroll
  for (int fi = 0; fi < 2; ++fi)
#pragma unroll
    for (int fj = 0; fj < 2; ++fj) {
      const int col = n0 + fj * 16 + lr;
      const float bv = bias ? bias[col] : 0.0f;
#pragma unroll
      for (int r = 0; r < 8; ++r) {
        const int row = m0 + fi * 16 + r + half * 8;   // C layout: row = r + 8*half
        float v = acc[fi][fj][r] + bv;
        if (act) v = 0.5f * v * (1.0f + erff(v * 0.70710678118654752f));
        const size_t idx = (size_t)row * N + col;
        if (C)  C[idx]  = v;
        if (Cb) Cb[idx] = f2bfu(v);
      }
    }
}

// ---------------------------------------------------------------------------
// Sliding-window flash attention on bf16 Q/K/V.
// One workgroup (512 threads = 16 waves) per (query-block, head, batch).
// Each wave owns a 16-query tile; K/V chunks (128 keys) staged to LDS with
// async global->LDS b128 copies (ASYNCcnt), V read back transposed.
// Out-of-window rows are clamped (their scores are masked to -1e9 -> exp==0).
// ---------------------------------------------------------------------------
__device__ __forceinline__ float mask_add(int ki, int qi, int qstart, int b,
                                          int S, const int* amask) {
  const int rel = ki - qi;                 // key window starts at q-block - 256
  const int kg  = qstart - 256 + ki;       // global key index
  const bool ok = (rel >= 0) && (rel <= 512) && (kg >= 0) && (kg < S);
  const int kgc = kg < 0 ? 0 : (kg >= S ? S - 1 : kg);
  const float km = (1.0f - (float)amask[(size_t)b * S + kgc]) * NEGV;
  return (ok ? 0.0f : NEGV) + km;
}

__global__ __launch_bounds__(512) void attn_kernel(
    const uint16_t* __restrict__ Q, const uint16_t* __restrict__ Kb,
    const uint16_t* __restrict__ Vb, const int* __restrict__ amask,
    uint16_t* __restrict__ Ctx, int S, int D) {
  const int nbq  = blockIdx.x;             // 0..15 query block
  const int head = blockIdx.y;             // 0..11
  const int b    = blockIdx.z;             // 0..1
  const int tid  = threadIdx.x;
  const int wave = tid >> 5, lane = tid & 31;
  const int lr = lane & 15, half = lane >> 4;
  const int qstart = nbq * 256;
  const int hoff   = head * 64;

  __shared__ uint16_t ks[128 * 64];        // K chunk, row-major key x dim (16 KB)
  __shared__ uint16_t vs[128 * 64];        // V chunk, row-major key x dim (16 KB)
  __shared__ uint16_t ps[16][16 * 32];     // per-wave P staging (16 KB)

  // This wave's Q tile (16 x 64) as two bf16 A-fragments straight from global.
  v16bf qf[2];
  qf[0] = ld_frag(Q, D, b * S + qstart + wave * 16, hoff, lane);
  qf[1] = ld_frag(Q, D, b * S + qstart + wave * 16, hoff + 32, lane);

  v8f acc[4] = {};                         // 16 x 64 output accumulator
  float mrow[8], lrow[8], fac[8];
#pragma unroll
  for (int r = 0; r < 8; ++r) { mrow[r] = -3.0e38f; lrow[r] = 0.0f; }

  const uint16_t* kbase = Kb + (size_t)(b * S) * D + hoff;
  const uint16_t* vbase = Vb + (size_t)(b * S) * D + hoff;

  for (int chunk = 0; chunk < 6; ++chunk) {   // 6 x 128 = 768-key window
    __syncthreads();
    // Stage 128 keys x 64 dims of K and V (128 B per row, 16 B per lane-op).
#pragma unroll
    for (int e = tid; e < 1024; e += 512) {
      const int j = e >> 3, sub = e & 7;
      const int kg = qstart - 256 + chunk * 128 + j;
      const int kgc = kg < 0 ? 0 : (kg >= S ? S - 1 : kg);
      const size_t go = (size_t)kgc * D + sub * 8;
      cp16_g2l(kbase + go, ks + j * 64 + sub * 8);
      cp16_g2l(vbase + go, vs + j * 64 + sub * 8);
    }
    async_drain();
    __syncthreads();

    for (int pp = 0; pp < 4; ++pp) {       // 32 keys per step
      const int kb0 = pp * 32;
      // scores: Q(16x64) @ K_tile^T -> two 16x16 f32 tiles
      v8f s0 = {}, s1 = {};
      {
        v16bf kf0a = ld_frag(ks, 64, kb0, 0, lane);
        v16bf kf0b = ld_frag(ks, 64, kb0, 32, lane);
        s0 = wmma_bf16(qf[0], kf0a, s0);
        s0 = wmma_bf16(qf[1], kf0b, s0);
        v16bf kf1a = ld_frag(ks, 64, kb0 + 16, 0, lane);
        v16bf kf1b = ld_frag(ks, 64, kb0 + 16, 32, lane);
        s1 = wmma_bf16(qf[0], kf1a, s1);
        s1 = wmma_bf16(qf[1], kf1b, s1);
      }
      uint16_t* pw = ps[wave];
#pragma unroll
      for (int r = 0; r < 8; ++r) {
        const int qi  = wave * 16 + r + half * 8;   // local query index
        const int ki0 = chunk * 128 + kb0 + lr;     // local key index in window
        const int ki1 = ki0 + 16;
        float sv0 = s0[r] * 0.125f + mask_add(ki0, qi, qstart, b, S, amask);
        float sv1 = s1[r] * 0.125f + mask_add(ki1, qi, qstart, b, S, amask);
        // online softmax: reduce across the 16-lane half-group
        float t = fmaxf(sv0, sv1);
#pragma unroll
        for (int o = 1; o < 16; o <<= 1) t = fmaxf(t, __shfl_xor(t, o, 32));
        const float mnew = fmaxf(mrow[r], t);
        const float f  = __expf(mrow[r] - mnew);
        const float p0 = __expf(sv0 - mnew);
        const float p1 = __expf(sv1 - mnew);
        float rs2 = p0 + p1;
#pragma unroll
        for (int o = 1; o < 16; o <<= 1) rs2 += __shfl_xor(rs2, o, 32);
        lrow[r] = lrow[r] * f + rs2;
        mrow[r] = mnew; fac[r] = f;
        const int rr = r + half * 8;                 // C-layout row
        pw[rr * 32 + lr]      = f2bfu(p0);
        pw[rr * 32 + 16 + lr] = f2bfu(p1);
      }
      // rescale running output, then O += P(16x32) @ V(32x64)
#pragma unroll
      for (int n = 0; n < 4; ++n)
#pragma unroll
        for (int r = 0; r < 8; ++r) acc[n][r] *= fac[r];
      v16bf pf = ld_frag(pw, 32, 0, 0, lane);
#pragma unroll
      for (int n = 0; n < 4; ++n) {
        v16bf vf = ld_frag_bT(vs, kb0, n * 16, lane);
        acc[n] = wmma_bf16(pf, vf, acc[n]);
      }
    }
  }

#pragma unroll
  for (int n = 0; n < 4; ++n)
#pragma unroll
    for (int r = 0; r < 8; ++r) {
      const int row = qstart + wave * 16 + r + half * 8;
      const int col = hoff + n * 16 + lr;
      Ctx[((size_t)(b * S + row)) * D + col] = f2bfu(acc[n][r] / lrow[r]);
    }
}

// ---------------------------------------------------------------------------
// Weight transpose + convert: W (K x N, f32) -> Wt (N x K, bf16)
// ---------------------------------------------------------------------------
__global__ void convert_wt_kernel(const float* __restrict__ W,
                                  uint16_t* __restrict__ Wt, int K, int N) {
  const size_t total = (size_t)K * N;
  for (size_t i = blockIdx.x * 256ull + threadIdx.x; i < total;
       i += (size_t)gridDim.x * 256) {
    const size_t n = i / K, k = i % K;
    Wt[i] = f2bfu(W[k * (size_t)N + n]);
  }
}

__global__ __launch_bounds__(256) void regress_kernel(
    const float* __restrict__ h, const float* __restrict__ rw,
    const float* __restrict__ rb, float* __restrict__ out, int S, int D) {
  const int b = blockIdx.x, tid = threadIdx.x;
  float s = 0.f;
  for (int d = tid; d < D; d += 256) s += h[(size_t)b * S * D + d] * rw[d];
  __shared__ float rs[256];
  rs[tid] = s; __syncthreads();
  for (int s2 = 128; s2 > 0; s2 >>= 1) {
    if (tid < s2) rs[tid] += rs[tid + s2];
    __syncthreads();
  }
  if (tid == 0) out[b] = rs[0] + rb[0];
}

// ---------------------------------------------------------------------------
// Host-side orchestration
// ---------------------------------------------------------------------------
extern "C" void kernel_launch(void* const* d_in, const int* in_sizes, int n_in,
                              void* d_out, int out_size, void* d_ws,
                              size_t ws_size, hipStream_t stream) {
  (void)in_sizes; (void)n_in; (void)out_size; (void)ws_size;
  const int Bc = 2, Sc = 4096, Dc = 768, Hc = 12, Fc = 3072;
  const int Mc = Bc * Sc;  // 8192 tokens

  const int*   ids  = (const int*)d_in[0];
  const int*   am   = (const int*)d_in[1];
  const float* wemb = (const float*)d_in[2];
  const float* pemb = (const float*)d_in[3];
  const float* temb = (const float*)d_in[4];
  const float* es   = (const float*)d_in[5];
  const float* ebv  = (const float*)d_in[6];
  const float* Wq   = (const float*)d_in[7];
  const float* bq   = (const float*)d_in[8];
  const float* Wk   = (const float*)d_in[9];
  const float* bk   = (const float*)d_in[10];
  const float* Wv   = (const float*)d_in[11];
  const float* bv   = (const float*)d_in[12];
  const float* Wo   = (const float*)d_in[13];
  const float* bo   = (const float*)d_in[14];
  const float* l1s  = (const float*)d_in[15];
  const float* l1b  = (const float*)d_in[16];
  const float* Wi   = (const float*)d_in[17];
  const float* bi   = (const float*)d_in[18];
  const float* Wf   = (const float*)d_in[19];
  const float* bfw  = (const float*)d_in[20];
  const float* l2s  = (const float*)d_in[21];
  const float* l2b  = (const float*)d_in[22];
  const float* rw   = (const float*)d_in[23];
  const float* rb   = (const float*)d_in[24];
  float* out = (float*)d_out;

  char* wsp = (char*)d_ws;
  size_t off = 0;
  auto alloc = [&](size_t bytes) -> void* {
    void* p = wsp + off;
    off += (bytes + 255) & ~(size_t)255;
    return p;
  };
  float*    h    = (float*)alloc((size_t)Mc * Dc * 4);
  uint16_t* hb   = (uint16_t*)alloc((size_t)Mc * Dc * 2);
  uint16_t* qbb  = (uint16_t*)alloc((size_t)Mc * Dc * 2);
  uint16_t* kbb  = (uint16_t*)alloc((size_t)Mc * Dc * 2);
  uint16_t* vbb  = (uint16_t*)alloc((size_t)Mc * Dc * 2);
  uint16_t* ctxb = (uint16_t*)alloc((size_t)Mc * Dc * 2);
  float*    tmp  = (float*)alloc((size_t)Mc * Dc * 4);
  float*    ff1  = (float*)alloc((size_t)Mc * Fc * 4);
  uint16_t* ff1b = (uint16_t*)alloc((size_t)Mc * Fc * 2);
  uint16_t* wqT  = (uint16_t*)alloc((size_t)Dc * Dc * 2);
  uint16_t* wkT  = (uint16_t*)alloc((size_t)Dc * Dc * 2);
  uint16_t* wvT  = (uint16_t*)alloc((size_t)Dc * Dc * 2);
  uint16_t* woT  = (uint16_t*)alloc((size_t)Dc * Dc * 2);
  uint16_t* wiT  = (uint16_t*)alloc((size_t)Dc * Fc * 2);
  uint16_t* wfT  = (uint16_t*)alloc((size_t)Fc * Dc * 2);

  embed_ln_kernel<<<Mc, 256, 0, stream>>>(ids, wemb, pemb, temb, es, ebv, h, hb,
                                          Sc, Dc);

  const dim3 gDD(Dc / 64, Mc / 128);   // N=768 GEMMs
  const dim3 gDF(Fc / 64, Mc / 128);   // N=3072 GEMM

  for (int l = 0; l < 2; ++l) {
    convert_wt_kernel<<<2048, 256, 0, stream>>>(Wq + (size_t)l * Dc * Dc, wqT, Dc, Dc);
    convert_wt_kernel<<<2048, 256, 0, stream>>>(Wk + (size_t)l * Dc * Dc, wkT, Dc, Dc);
    convert_wt_kernel<<<2048, 256, 0, stream>>>(Wv + (size_t)l * Dc * Dc, wvT, Dc, Dc);
    convert_wt_kernel<<<2048, 256, 0, stream>>>(Wo + (size_t)l * Dc * Dc, woT, Dc, Dc);
    convert_wt_kernel<<<2048, 256, 0, stream>>>(Wi + (size_t)l * Dc * Fc, wiT, Dc, Fc);
    convert_wt_kernel<<<2048, 256, 0, stream>>>(Wf + (size_t)l * Fc * Dc, wfT, Fc, Dc);

    // Q/K/V: bf16-only outputs (consumed by attention)
    gemm_kernel<<<gDD, 256, 0, stream>>>(hb, wqT, bq + (size_t)l * Dc, nullptr,
                                         qbb, Mc, Dc, Dc, 0);
    gemm_kernel<<<gDD, 256, 0, stream>>>(hb, wkT, bk + (size_t)l * Dc, nullptr,
                                         kbb, Mc, Dc, Dc, 0);
    gemm_kernel<<<gDD, 256, 0, stream>>>(hb, wvT, bv + (size_t)l * Dc, nullptr,
                                         vbb, Mc, Dc, Dc, 0);

    attn_kernel<<<dim3(Sc / 256, Hc, Bc), 512, 0, stream>>>(qbb, kbb, vbb, am,
                                                            ctxb, Sc, Dc);

    gemm_kernel<<<gDD, 256, 0, stream>>>(ctxb, woT, bo + (size_t)l * Dc, tmp,
                                         nullptr, Mc, Dc, Dc, 0);
    add_ln_kernel<<<Mc, 256, 0, stream>>>(h, tmp, l1s + (size_t)l * Dc,
                                          l1b + (size_t)l * Dc, h, hb, Dc);

    gemm_kernel<<<gDF, 256, 0, stream>>>(hb, wiT, bi + (size_t)l * Fc, ff1, ff1b,
                                         Mc, Fc, Dc, 1);
    gemm_kernel<<<gDD, 256, 0, stream>>>(ff1b, wfT, bfw + (size_t)l * Dc, tmp,
                                         nullptr, Mc, Dc, Fc, 0);
    add_ln_kernel<<<Mc, 256, 0, stream>>>(h, tmp, l2s + (size_t)l * Dc,
                                          l2b + (size_t)l * Dc, h, hb, Dc);
  }

  regress_kernel<<<Bc, 256, 0, stream>>>(h, rw, rb, out, Sc, Dc);
}
